// NormLinearAttention_23948737643218
// MI455X (gfx1250) — compile-verified, matrix operand order checked by reading, not run on hardware
//
#include <hip/hip_runtime.h>

// ---- problem constants (match reference) ----
#define BB    8
#define NN    8192
#define DIMD  512
#define HH    8
#define DHD   64
#define INNERD 512            // HH*DHD
#define MTOT  (BB * NN)       // 65536
#define BHN   (BB * HH)       // 64

typedef __attribute__((ext_vector_type(16))) __bf16 v16bf;
typedef __attribute__((ext_vector_type(8)))  __bf16 v8bf;
typedef __attribute__((ext_vector_type(8)))  float  v8f;

// ---------------------------------------------------------------------------
// WMMA helpers.  CDNA5 wave32 bf16 WMMA: D(16x16,f32) = A(16x32) * B(32x16) + C
// A fragment (per lane): row = (lane&15), K indices:
//   lanes 0-15 : K = {k0..k0+7, k0+16..k0+23}
//   lanes 16-31: K = {k0+8..k0+15, k0+24..k0+31}
// B fragment is the dual layout on a [N][K] array -> same loader.
// ---------------------------------------------------------------------------
__device__ __forceinline__ v16bf load_frag(const __bf16* __restrict__ mat,
                                           int ld, int idx, int k0, int lane) {
  const __bf16* p = mat + (size_t)idx * (size_t)ld + k0 + ((lane & 16) ? 8 : 0);
  v8bf lo = *(const v8bf*)(p);
  v8bf hi = *(const v8bf*)(p + 16);
  v16bf f;
#pragma unroll
  for (int i = 0; i < 8; ++i) { f[i] = lo[i]; f[i + 8] = hi[i]; }
  return f;
}

// Fragment from an LDS-staged [64 rows][32 halfs] tile (row = B column index).
__device__ __forceinline__ v16bf lds_frag(const __bf16* sm, int row, int lane) {
  const __bf16* p = sm + row * 32 + ((lane & 16) ? 8 : 0);
  v8bf lo = *(const v8bf*)(p);
  v8bf hi = *(const v8bf*)(p + 16);
  v16bf f;
#pragma unroll
  for (int i = 0; i < 8; ++i) { f[i] = lo[i]; f[i + 8] = hi[i]; }
  return f;
}

__device__ __forceinline__ v8f wmma_bf16(v16bf a, v16bf b, v8f c) {
  return __builtin_amdgcn_wmma_f32_16x16x32_bf16(false, a, false, b,
                                                 (short)0, c, false, false);
}

// gfx1250 async copy global -> LDS (16B per lane), tracked with ASYNCcnt.
__device__ __forceinline__ void async_load_b128(unsigned lds_off,
                                                const __bf16* gptr) {
  asm volatile("global_load_async_to_lds_b128 %0, %1, off"
               :: "v"(lds_off), "v"(gptr)
               : "memory");
}

__device__ __forceinline__ void wait_async0() {
  asm volatile("s_wait_asynccnt 0" ::: "memory");
}

// ---------------------------------------------------------------------------
// Conversion / utility kernels
// ---------------------------------------------------------------------------
__global__ void cvt_f32_bf16(const float* __restrict__ src,
                             __bf16* __restrict__ dst, size_t n) {
  size_t i = (size_t)blockIdx.x * blockDim.x + threadIdx.x;
  if (i < n) dst[i] = (__bf16)src[i];
}

// src [rows][cols] f32  ->  dst [cols][rows] bf16
__global__ void transpose_cvt(const float* __restrict__ src,
                              __bf16* __restrict__ dst, int rows, int cols) {
  int i = blockIdx.x * blockDim.x + threadIdx.x;
  if (i < rows * cols) {
    int r = i / cols, c = i - r * cols;
    dst[(size_t)c * rows + r] = (__bf16)src[i];
  }
}

__global__ void zero_f32(float* __restrict__ p, size_t n) {
  size_t i = (size_t)blockIdx.x * blockDim.x + threadIdx.x;
  if (i < n) p[i] = 0.0f;
}

// kvf [bh][m][d] f32 -> kvT [bh][d][m] bf16   (4096 elems per bh)
__global__ void kv_transpose(const float* __restrict__ kvf,
                             __bf16* __restrict__ kvT) {
  int i = blockIdx.x * blockDim.x + threadIdx.x;
  if (i < BHN * DHD * DHD) {
    int bh = i >> 12;
    int rem = i & 4095;
    int m = rem >> 6, d = rem & 63;
    kvT[(bh << 12) + (d << 6) + m] = (__bf16)kvf[i];
  }
}

// ---------------------------------------------------------------------------
// Stage 1: qkv = x @ w_qkv  (M=65536, K=512, N=1536), relu(q), relu(k),
// scatter q -> [bh][n][d], k,v -> [bh][d][n] (transposed for next stage).
// Block = 256 threads = 8 waves; wave computes 16(M) x 64(N); block 128x64.
// B tile (64 cols x 32 K = 4KB) staged in LDS via async copy, double-buffered.
// ---------------------------------------------------------------------------
__global__ __launch_bounds__(256) void gemm_qkv(
    const __bf16* __restrict__ xb, const __bf16* __restrict__ wT,
    __bf16* __restrict__ qb, __bf16* __restrict__ kbt, __bf16* __restrict__ vbt) {
  __shared__ __align__(16) __bf16 smB[2][64 * 32];

  int lane = threadIdx.x & 31;
  int wave = threadIdx.x >> 5;
  int r15 = lane & 15;
  int m0 = blockIdx.x * 128 + wave * 16;
  int n0 = blockIdx.y * 64;

  // async-fill assignment: thread t moves 16B of the 64x32 B tile
  int t = threadIdx.x;
  int brow = t >> 2;               // 0..63 (B column index)
  int bseg = (t & 3) * 8;          // half-offset within the 32-K chunk
  const __bf16* bsrc = wT + (size_t)(n0 + brow) * DIMD + bseg;
  unsigned smb[2];
  smb[0] = (unsigned)(unsigned long long)&smB[0][brow * 32 + bseg];
  smb[1] = (unsigned)(unsigned long long)&smB[1][brow * 32 + bseg];

  async_load_b128(smb[0], bsrc);   // preload K-chunk 0

  v8f acc[4] = {};
  v16bf a = load_frag(xb, DIMD, m0 + r15, 0, lane);
  for (int kc = 0; kc < 16; ++kc) {
    int k = kc * 32;
    wait_async0();
    __syncthreads();               // buf[kc&1] ready; buf[(kc+1)&1] free
    if (kc + 1 < 16)
      async_load_b128(smb[(kc + 1) & 1], bsrc + (k + 32));
    v16bf a_n = a;
    if (kc + 1 < 16) {
      __builtin_prefetch(xb + (size_t)(m0 + r15) * DIMD + k + 64, 0, 1);
      a_n = load_frag(xb, DIMD, m0 + r15, k + 32, lane);
    }
    const __bf16* smcur = smB[kc & 1];
#pragma unroll
    for (int j = 0; j < 4; ++j) {
      v16bf b = lds_frag(smcur, j * 16 + r15, lane);
      acc[j] = wmma_bf16(a, b, acc[j]);
    }
    a = a_n;
  }

  int rowadd = (lane < 16) ? 0 : 8;
#pragma unroll
  for (int j = 0; j < 4; ++j) {
#pragma unroll
    for (int r = 0; r < 8; ++r) {
      int row = m0 + rowadd + r;             // global token row
      int col = n0 + j * 16 + r15;           // column in [0,1536)
      float v = acc[j][r];
      int b  = row >> 13;                    // / 8192
      int n  = row & (NN - 1);
      int sec = col >> 9;                    // 0=q 1=k 2=v
      int c  = col & (INNERD - 1);
      int h  = c >> 6, d = c & 63;
      int bh = b * HH + h;
      if (sec == 0) {
        v = v > 0.0f ? v : 0.0f;
        qb[((size_t)bh * NN + n) * DHD + d] = (__bf16)v;
      } else if (sec == 1) {
        v = v > 0.0f ? v : 0.0f;
        kbt[((size_t)bh * DHD + d) * NN + n] = (__bf16)v;
      } else {
        vbt[((size_t)bh * DHD + d) * NN + n] = (__bf16)v;
      }
    }
  }
}

// ---------------------------------------------------------------------------
// Stage 2: kv[bh][m][d] = sum_n kbt[bh][m][n] * vbt[bh][d][n]   (K^T V)
// grid = (BHN, 8 K-splits); block 256 = 8 waves; wave = 16(m) x 32(d).
// ---------------------------------------------------------------------------
__global__ __launch_bounds__(256) void kv_accum(
    const __bf16* __restrict__ kbt, const __bf16* __restrict__ vbt,
    float* __restrict__ kvf) {
  int bh = blockIdx.x;
  int ks = blockIdx.y;
  int lane = threadIdx.x & 31;
  int wave = threadIdx.x >> 5;
  int r15 = lane & 15;
  int m0 = (wave >> 1) * 16;
  int d0 = (wave & 1) * 32;
  const __bf16* A  = kbt + (size_t)bh * DHD * NN;
  const __bf16* Bm = vbt + (size_t)bh * DHD * NN;

  v8f acc[2] = {};
  int kbeg = ks * (NN / 8), kend = kbeg + (NN / 8);
  for (int k = kbeg; k < kend; k += 32) {
    v16bf a = load_frag(A, NN, m0 + r15, k, lane);
#pragma unroll
    for (int j = 0; j < 2; ++j) {
      v16bf b = load_frag(Bm, NN, d0 + j * 16 + r15, k, lane);
      acc[j] = wmma_bf16(a, b, acc[j]);
    }
  }

  int rowadd = (lane < 16) ? 0 : 8;
#pragma unroll
  for (int j = 0; j < 2; ++j) {
#pragma unroll
    for (int r = 0; r < 8; ++r) {
      int m = m0 + rowadd + r;
      int d = d0 + j * 16 + r15;
      atomicAdd(&kvf[((size_t)bh * DHD + m) * DHD + d], acc[j][r]);
    }
  }
}

// ---------------------------------------------------------------------------
// Stage 3: out[bh][n][d] = sum_m q[bh][n][m] * kv[bh][m][d]   (K = 64)
// writes concatenated-head layout attn[(b*NN+n)][h*64+d] (bf16).
// ---------------------------------------------------------------------------
__global__ __launch_bounds__(256) void attn_out(
    const __bf16* __restrict__ qb, const __bf16* __restrict__ kvT,
    __bf16* __restrict__ attn) {
  int bh = blockIdx.y;
  int b = bh >> 3, h = bh & 7;
  int lane = threadIdx.x & 31;
  int wave = threadIdx.x >> 5;
  int r15 = lane & 15;
  int n0 = blockIdx.x * 128 + wave * 16;
  const __bf16* A  = qb  + (size_t)bh * NN * DHD;
  const __bf16* Bm = kvT + (size_t)bh * DHD * DHD;

  v8f acc[4] = {};
#pragma unroll
  for (int k = 0; k < DHD; k += 32) {
    v16bf a = load_frag(A, DHD, n0 + r15, k, lane);
#pragma unroll
    for (int j = 0; j < 4; ++j) {
      v16bf bfr = load_frag(Bm, DHD, j * 16 + r15, k, lane);
      acc[j] = wmma_bf16(a, bfr, acc[j]);
    }
  }

  int rowadd = (lane < 16) ? 0 : 8;
#pragma unroll
  for (int j = 0; j < 4; ++j) {
#pragma unroll
    for (int r = 0; r < 8; ++r) {
      int n = n0 + rowadd + r;
      int col = j * 16 + r15;
      attn[((size_t)(b * NN + n)) * INNERD + h * DHD + col] = (__bf16)acc[j][r];
    }
  }
}

// ---------------------------------------------------------------------------
// Stage 4: row LayerNorm over INNER=512 (one wave per row; 16 elems/lane).
// ---------------------------------------------------------------------------
__global__ __launch_bounds__(256) void layernorm_rows(
    const __bf16* __restrict__ attn, const float* __restrict__ lnw,
    const float* __restrict__ lnb, __bf16* __restrict__ ynorm) {
  int lane = threadIdx.x & 31;
  int wave = threadIdx.x >> 5;
  size_t row = (size_t)blockIdx.x * 8 + wave;
  const __bf16* p = attn + row * INNERD + lane * 16;
  v8bf a = *(const v8bf*)(p);
  v8bf b = *(const v8bf*)(p + 8);
  float vals[16];
#pragma unroll
  for (int i = 0; i < 8; ++i) { vals[i] = (float)a[i]; vals[8 + i] = (float)b[i]; }
  float s = 0.0f, s2 = 0.0f;
#pragma unroll
  for (int i = 0; i < 16; ++i) { s += vals[i]; s2 += vals[i] * vals[i]; }
#pragma unroll
  for (int off = 16; off >= 1; off >>= 1) {
    s  += __shfl_xor(s,  off, 32);
    s2 += __shfl_xor(s2, off, 32);
  }
  float mu  = s * (1.0f / INNERD);
  float var = s2 * (1.0f / INNERD) - mu * mu;
  float inv = rsqrtf(var + 1e-5f);
  int c0 = lane * 16;
  v8bf lo, hi;
#pragma unroll
  for (int i = 0; i < 8; ++i) {
    lo[i] = (__bf16)((vals[i]     - mu) * inv * lnw[c0 + i]     + lnb[c0 + i]);
    hi[i] = (__bf16)((vals[8 + i] - mu) * inv * lnw[c0 + 8 + i] + lnb[c0 + 8 + i]);
  }
  *(v8bf*)(ynorm + row * INNERD + c0)     = lo;
  *(v8bf*)(ynorm + row * INNERD + c0 + 8) = hi;
}

// ---------------------------------------------------------------------------
// Stage 5: out = ynorm @ w_out + b_out   (M=65536, K=512, N=512), f32 out.
// Same async-LDS double-buffered B staging as gemm_qkv.
// ---------------------------------------------------------------------------
__global__ __launch_bounds__(256) void gemm_out(
    const __bf16* __restrict__ ynorm, const __bf16* __restrict__ woutT,
    const float* __restrict__ bout, float* __restrict__ out) {
  __shared__ __align__(16) __bf16 smB[2][64 * 32];

  int lane = threadIdx.x & 31;
  int wave = threadIdx.x >> 5;
  int r15 = lane & 15;
  int m0 = blockIdx.x * 128 + wave * 16;
  int n0 = blockIdx.y * 64;

  int t = threadIdx.x;
  int brow = t >> 2;
  int bseg = (t & 3) * 8;
  const __bf16* bsrc = woutT + (size_t)(n0 + brow) * DIMD + bseg;
  unsigned smb[2];
  smb[0] = (unsigned)(unsigned long long)&smB[0][brow * 32 + bseg];
  smb[1] = (unsigned)(unsigned long long)&smB[1][brow * 32 + bseg];

  async_load_b128(smb[0], bsrc);

  v8f acc[4] = {};
  v16bf a = load_frag(ynorm, DIMD, m0 + r15, 0, lane);
  for (int kc = 0; kc < 16; ++kc) {
    int k = kc * 32;
    wait_async0();
    __syncthreads();
    if (kc + 1 < 16)
      async_load_b128(smb[(kc + 1) & 1], bsrc + (k + 32));
    v16bf a_n = a;
    if (kc + 1 < 16) {
      __builtin_prefetch(ynorm + (size_t)(m0 + r15) * DIMD + k + 64, 0, 1);
      a_n = load_frag(ynorm, DIMD, m0 + r15, k + 32, lane);
    }
    const __bf16* smcur = smB[kc & 1];
#pragma unroll
    for (int j = 0; j < 4; ++j) {
      v16bf b = lds_frag(smcur, j * 16 + r15, lane);
      acc[j] = wmma_bf16(a, b, acc[j]);
    }
    a = a_n;
  }

  int rowadd = (lane < 16) ? 0 : 8;
#pragma unroll
  for (int j = 0; j < 4; ++j) {
    int col = n0 + j * 16 + r15;
    float bj = bout[col];
#pragma unroll
    for (int r = 0; r < 8; ++r) {
      int row = m0 + rowadd + r;
      out[(size_t)row * DIMD + col] = acc[j][r] + bj;
    }
  }
}

// ---------------------------------------------------------------------------
// Host-side launch
// ---------------------------------------------------------------------------
extern "C" void kernel_launch(void* const* d_in, const int* in_sizes, int n_in,
                              void* d_out, int out_size, void* d_ws, size_t ws_size,
                              hipStream_t stream) {
  const float* x     = (const float*)d_in[0];
  const float* w_qkv = (const float*)d_in[1];
  const float* ln_w  = (const float*)d_in[2];
  const float* ln_b  = (const float*)d_in[3];
  const float* w_out = (const float*)d_in[4];
  const float* b_out = (const float*)d_in[5];
  float* out = (float*)d_out;

  char* ws = (char*)d_ws;
  // workspace layout (bytes), all 256-aligned
  __bf16* xb    = (__bf16*)(ws + 0);              // 65536*512*2   = 64 MiB
  __bf16* wqkvT = (__bf16*)(ws + 67108864);       // 1536*512*2
  __bf16* woutT = (__bf16*)(ws + 68681728);       // 512*512*2
  __bf16* qb    = (__bf16*)(ws + 69206016);       // [bh][n][d]    = 64 MiB
  __bf16* kbt   = (__bf16*)(ws + 136314880);      // [bh][d][n]    = 64 MiB
  __bf16* vbt   = (__bf16*)(ws + 203423744);      // [bh][d][n]    = 64 MiB
  float*  kvf   = (float*) (ws + 270532608);      // [bh][64][64] f32
  __bf16* kvT   = (__bf16*)(ws + 271581184);      // [bh][d][m] bf16
  __bf16* attn  = kbt;    // reuse (kbt dead after kv_accum)
  __bf16* ynorm = xb;     // reuse (xb dead after gemm_qkv)

  const size_t nx = (size_t)MTOT * DIMD;          // 33,554,432

  cvt_f32_bf16<<<dim3((unsigned)((nx + 255) / 256)), dim3(256), 0, stream>>>(x, xb, nx);
  transpose_cvt<<<dim3((512 * 1536 + 255) / 256), dim3(256), 0, stream>>>(w_qkv, wqkvT, 512, 1536);
  transpose_cvt<<<dim3((512 * 512 + 255) / 256), dim3(256), 0, stream>>>(w_out, woutT, 512, 512);
  zero_f32<<<dim3((BHN * DHD * DHD + 255) / 256), dim3(256), 0, stream>>>(kvf, (size_t)BHN * DHD * DHD);

  gemm_qkv<<<dim3(MTOT / 128, (3 * INNERD) / 64), dim3(256), 0, stream>>>(xb, wqkvT, qb, kbt, vbt);

  kv_accum<<<dim3(BHN, 8), dim3(256), 0, stream>>>(kbt, vbt, kvf);
  kv_transpose<<<dim3((BHN * DHD * DHD + 255) / 256), dim3(256), 0, stream>>>(kvf, kvT);

  attn_out<<<dim3(NN / 128, BHN), dim3(256), 0, stream>>>(qb, kvT, attn);

  layernorm_rows<<<dim3(MTOT / 8), dim3(256), 0, stream>>>(attn, ln_w, ln_b, ynorm);

  gemm_out<<<dim3(MTOT / 128, DIMD / 64), dim3(256), 0, stream>>>(ynorm, woutT, b_out, out);
}
